// lstm_crf_11553462026311
// MI455X (gfx1250) — matrix operand order repaired and checked by reading, not compiled, not run on hardware
//
#include <hip/hip_runtime.h>
#include <math.h>

typedef __attribute__((ext_vector_type(16))) _Float16 v16h;
typedef __attribute__((ext_vector_type(8)))  float    v8f;

#define BATCH 64
#define TLEN  512
#define EMB   300
#define HALF  150
#define NP    640          // packed gate dim: n' = 4*j + g, j padded to 160
#define KPX   320          // padded K for x-projection (300 -> 320)
#define KPH   160          // padded K for h-projection (150 -> 160)
#define NTAGS 16
#define START_TAG 14
#define STOP_TAG  15
#define NEGV  (-10000.0f)

#if __has_builtin(__builtin_amdgcn_sched_barrier)
#define SCHED_FENCE() __builtin_amdgcn_sched_barrier(0)
#else
#define SCHED_FENCE()
#endif

// ---------------- WMMA fragment helpers (CDNA5 wave32 layouts) ----------------

// A-matrix 16x32 f16: lane l -> M = l&15 ; half=l>>4 ; element e -> K = 16*half + e
__device__ __forceinline__ v16h load_a_frag(const _Float16* base, int row_stride,
                                            int mtile, int lane, int kbase) {
  v16h a;
  int m    = mtile * 16 + (lane & 15);
  int half = lane >> 4;
  const _Float16* p = base + (size_t)m * row_stride + kbase;
#pragma unroll
  for (int e = 0; e < 16; ++e) {
    int v  = e >> 1;
    int kk = 16 * half + e; (void)v;
    a[e] = p[kk];
  }
  return a;
}

// B-matrix 32x16 f16 (row-major N x K source): lane l -> N = l&15 ; K = 16*(l>>4)+e
__device__ __forceinline__ v16h load_b_frag(const _Float16* base, int row_stride,
                                            int ntile, int lane, int kbase) {
  v16h b;
  int n = ntile * 16 + (lane & 15);
  const _Float16* p = base + (size_t)n * row_stride + kbase + ((lane >> 4) * 16);
#pragma unroll
  for (int e = 0; e < 16; ++e) b[e] = p[e];
  return b;
}

__device__ __forceinline__ v8f wmma_f16(v16h a, v16h b, v8f c) {
  return __builtin_amdgcn_wmma_f32_16x16x32_f16(false, a, false, b, (short)0, c,
                                                false, false);
}

__device__ __forceinline__ float sigm(float x) { return 1.0f / (1.0f + expf(-x)); }

// ---------------- prep kernels ----------------

__global__ void zero_half_kernel(_Float16* p, long n) {
  long i = (long)blockIdx.x * blockDim.x + threadIdx.x;
  if (i < n) p[i] = (_Float16)0.0f;
}

// pack W[(2, 600, K)] -> dst[(2, 640, Kpad)] with n' = 4*j + g interleave, zero pad
__global__ void pack_w_gates_kernel(const float* src, _Float16* dst, int K, int Kpad,
                                    int total) {
  int idx = blockIdx.x * blockDim.x + threadIdx.x;
  if (idx >= total) return;
  int k  = idx % Kpad;
  int np = (idx / Kpad) % NP;
  int d  = idx / (Kpad * NP);
  int j = np >> 2, g = np & 3;
  float v = 0.0f;
  if (j < HALF && k < K) v = src[((size_t)d * 600 + g * HALF + j) * K + k];
  dst[idx] = (_Float16)v;
}

// bias: dst[(2,640)] = bih + bhh, interleaved, zero pad
__global__ void pack_b_gates_kernel(const float* bih, const float* bhh, float* dst) {
  int idx = blockIdx.x * blockDim.x + threadIdx.x;
  if (idx >= 2 * NP) return;
  int np = idx % NP, d = idx / NP;
  int j = np >> 2, g = np & 3;
  float v = 0.0f;
  if (j < HALF) v = bih[d * 600 + g * HALF + j] + bhh[d * 600 + g * HALF + j];
  dst[idx] = v;
}

__global__ void pack_wout_kernel(const float* src, _Float16* dst) {
  int idx = blockIdx.x * blockDim.x + threadIdx.x;
  if (idx >= NTAGS * KPX) return;
  int k = idx % KPX, n = idx / KPX;
  float v = (k < EMB) ? src[n * EMB + k] : 0.0f;
  dst[idx] = (_Float16)v;
}

// embedding gather -> e_h[(T,64,320)] fp16, plus mask[(64,512)] f32
__global__ void embed_mask_kernel(const int* x, const float* embed, _Float16* e_h,
                                  float* mask) {
  long idx = (long)blockIdx.x * blockDim.x + threadIdx.x;
  if (idx >= (long)TLEN * BATCH * KPX) return;
  int k = (int)(idx % KPX);
  long r = idx / KPX;
  int b = (int)(r % BATCH);
  int t = (int)(r / BATCH);
  int tok = x[b * TLEN + t];
  float v = (k < EMB) ? embed[(size_t)tok * EMB + k] : 0.0f;
  e_h[idx] = (_Float16)v;
  if (k == 0) mask[b * TLEN + t] = (tok > 0) ? 1.0f : 0.0f;
}

// ---------------- parallel input-projection GEMM (bias folded in) ----------------
// X:(T,64,320) fp16  Wp:(2,640,320) fp16  biasp:(2,640)  XP:(2,T,64,640) f32
__global__ void __launch_bounds__(256, 1)
xproj_kernel(const _Float16* X, const _Float16* Wp, const float* biasp, float* XP) {
  int t   = blockIdx.x;
  int dir = blockIdx.y;
  int w = threadIdx.x >> 5, l = threadIdx.x & 31;
  int mtile = w >> 1;
  int nstart = (w & 1) * 20;
  const _Float16* Xt = X + (size_t)t * BATCH * KPX;
  const _Float16* Wd = Wp + (size_t)dir * NP * KPX;
  v16h af[10];
#pragma unroll
  for (int kt = 0; kt < 10; ++kt) af[kt] = load_a_frag(Xt, KPX, mtile, l, kt * 32);
  for (int nt = 0; nt < 20; ++nt) {
    int ntile = nstart + nt;
    int ncol = ntile * 16 + (l & 15);
    float bv = biasp[dir * NP + ncol];
    v8f acc;
#pragma unroll
    for (int v = 0; v < 8; ++v) acc[v] = bv;
    // batch ALL B-fragment loads, fence the scheduler, then run the WMMA chain:
    // 20 loads stay in flight -> one exposed L2 latency per tile instead of ten
    v16h bf[10];
#pragma unroll
    for (int kt = 0; kt < 10; ++kt) bf[kt] = load_b_frag(Wd, KPX, ntile, l, kt * 32);
    SCHED_FENCE();
#pragma unroll
    for (int kt = 0; kt < 10; ++kt) acc = wmma_f16(af[kt], bf[kt], acc);
    size_t base = (((size_t)dir * TLEN + t) * BATCH) * NP;
#pragma unroll
    for (int v = 0; v < 8; ++v) {
      int m = mtile * 16 + 8 * (l >> 4) + v;
      XP[base + (size_t)m * NP + ncol] = acc[v];
    }
  }
}

// ---------------- persistent recurrent LSTM kernel (one block per direction) ----
// XP:(2,T,64,640) f32 (x-proj + bias), Whhp:(2,640,160) fp16, hout:(T,64,320) fp16
// Dynamic LDS: this direction's Whh slab (640*160 fp16 = 200 KB; WGP has 320 KB)
__global__ void __launch_bounds__(256, 1)
lstm_kernel(const float* XP, const _Float16* Whhp, const float* mask,
            _Float16* hout) {
  __shared__ _Float16 h_sh[BATCH][KPH];   // 20.0 KB
  __shared__ float    c_sh[BATCH][KPH];   // 40.0 KB
  extern __shared__ _Float16 wsh[];       // 640*160 fp16 = 200 KB (dynamic)
  int dir = blockIdx.x;
  int tid = threadIdx.x;
  int w = tid >> 5, l = tid & 31;
  int mtile = w >> 1;
  int nstart = (w & 1) * 20;

  for (int i = tid; i < BATCH * KPH; i += 256) {
    ((_Float16*)h_sh)[i] = (_Float16)0.0f;
    ((float*)c_sh)[i] = 0.0f;
  }
  // stage Whh (this direction) into LDS: 204800 B as uint4 chunks
  {
    const uint4* src = (const uint4*)(Whhp + (size_t)dir * NP * KPH);
    uint4* dst = (uint4*)wsh;
    for (int i = tid; i < (NP * KPH * 2) / 16; i += 256) dst[i] = src[i];
  }
  __syncthreads();

  for (int tt = 0; tt < TLEN; ++tt) {
    int t = dir ? (TLEN - 1 - tt) : tt;
    // phase 1: snapshot h into A fragments (registers)
    v16h af[5];
#pragma unroll
    for (int kt = 0; kt < 5; ++kt)
      af[kt] = load_a_frag(&h_sh[0][0], KPH, mtile, l, kt * 32);

    // prefetch next timestep's gate block (64x640 f32 = 160 KB) into caches
    {
      int tn = dir ? (t - 1) : (t + 1);
      if (tn >= 0 && tn < TLEN) {
        const char* pn =
            (const char*)(XP + (((size_t)dir * TLEN + tn) * BATCH) * NP);
#pragma unroll
        for (int i = 0; i < 5; ++i)
          __builtin_prefetch(pn + ((size_t)(i * 256 + tid)) * 128, 0, 1);
      }
    }
    __syncthreads();

    const float* XPt = XP + (((size_t)dir * TLEN + t) * BATCH) * NP;
    // per-row masks for this step (hoisted out of the tile loop)
    float mrow[8];
#pragma unroll
    for (int v = 0; v < 8; ++v)
      mrow[v] = mask[(mtile * 16 + 8 * (l >> 4) + v) * TLEN + t];

    for (int nt = 0; nt < 20; ++nt) {
      int ntile = nstart + nt;
      int ncol = ntile * 16 + (l & 15);
      v8f acc;
#pragma unroll
      for (int v = 0; v < 8; ++v) {
        int m = mtile * 16 + 8 * (l >> 4) + v;
        acc[v] = XPt[(size_t)m * NP + ncol];
      }
      // batch all 5 B fragments from LDS + acc-init loads, then fence, then WMMAs
      v16h bf[5];
#pragma unroll
      for (int kt = 0; kt < 5; ++kt) {
        const _Float16* pw = &wsh[(size_t)ncol * KPH + kt * 32 + (l >> 4) * 16];
#pragma unroll
        for (int e = 0; e < 16; ++e) bf[kt][e] = pw[e];
      }
      SCHED_FENCE();
#pragma unroll
      for (int kt = 0; kt < 5; ++kt) acc = wmma_f16(af[kt], bf[kt], acc);

      // elementwise LSTM: n' = 4*j + g  ->  i,f,g,o live in quad lanes
      int j = ntile * 4 + ((l & 15) >> 2);
      int qbase = l & ~3;
#pragma unroll
      for (int v = 0; v < 8; ++v) {
        float xi = __shfl(acc[v], qbase + 0, 32);
        float xf = __shfl(acc[v], qbase + 1, 32);
        float xg = __shfl(acc[v], qbase + 2, 32);
        float xo = __shfl(acc[v], qbase + 3, 32);
        int m = mtile * 16 + 8 * (l >> 4) + v;
        float c_old = c_sh[m][j];
        float h_old = (float)h_sh[m][j];
        float cn = sigm(xf) * c_old + sigm(xi) * tanhf(xg);
        float hn = sigm(xo) * tanhf(cn);
        float mt = mrow[v];
        float hu = mt * hn + (1.0f - mt) * h_old;
        float cu = mt * cn + (1.0f - mt) * c_old;
        if ((l & 3) == 0) {
          c_sh[m][j] = cu;
          h_sh[m][j] = (_Float16)hu;
          if (j < HALF)
            hout[((size_t)t * BATCH + m) * KPX + dir * HALF + j] = (_Float16)hu;
        }
      }
    }
    __syncthreads();
  }
}

// ---------------- emission projection: (32768 x 320) x (320 x 16) -----------
__global__ void __launch_bounds__(256, 1)
emis_kernel(const _Float16* h2, const _Float16* Wo, const float* bout,
            const float* mask, float* emis) {
  int w = threadIdx.x >> 5, l = threadIdx.x & 31;
  int mtile = blockIdx.x * 8 + w;  // 0..2047
  float bv = bout[l & 15];
  v8f acc;
#pragma unroll
  for (int v = 0; v < 8; ++v) acc[v] = bv;
  // preload all fragments, fence, then run the WMMA chain
  v16h af[10], bf[10];
#pragma unroll
  for (int kt = 0; kt < 10; ++kt) {
    af[kt] = load_a_frag(h2, KPX, mtile, l, kt * 32);
    bf[kt] = load_b_frag(Wo, KPX, 0, l, kt * 32);
  }
  SCHED_FENCE();
#pragma unroll
  for (int kt = 0; kt < 10; ++kt) acc = wmma_f16(af[kt], bf[kt], acc);
#pragma unroll
  for (int v = 0; v < 8; ++v) {
    int r = mtile * 16 + 8 * (l >> 4) + v;   // r = t*64 + b
    int t = r >> 6, b = r & 63;
    float mt = mask[b * TLEN + t];
    emis[((size_t)b * TLEN + t) * NTAGS + (l & 15)] = acc[v] * mt;
  }
}

// ---------------- CRF partition + score, one wave per batch row --------------
__global__ void crf_kernel(const float* emis, const float* mask, const float* trans,
                           const int* y, float* out) {
  __shared__ float tr[NTAGS * NTAGS];
  int b = blockIdx.x;
  int l = threadIdx.x;            // 0..31
  int ii = l & 15;
  for (int i = l; i < NTAGS * NTAGS; i += 32) tr[i] = trans[i];
  __syncthreads();

  float alpha = (ii == START_TAG) ? 0.0f : NEGV;
  for (int t = 0; t < TLEN; ++t) {
    float ht = emis[((size_t)b * TLEN + t) * NTAGS + ii];
    float mx = -3.0e38f;
#pragma unroll
    for (int j = 0; j < NTAGS; ++j) {
      float v = __shfl(alpha, j, 32) + tr[ii * NTAGS + j];
      mx = fmaxf(mx, v);
    }
    float s = 0.0f;
#pragma unroll
    for (int j = 0; j < NTAGS; ++j) {
      float v = __shfl(alpha, j, 32) + tr[ii * NTAGS + j];
      s += expf(v - mx);
    }
    float anew = mx + logf(s) + ht;
    float mt = mask[b * TLEN + t];
    alpha = mt * anew + (1.0f - mt) * alpha;
  }
  // Z = logsumexp_j(alpha[j] + trans[STOP][j]) ; reduce within each 16-lane half
  float zv = alpha + tr[STOP_TAG * NTAGS + ii];
  float zmax = zv;
#pragma unroll
  for (int m = 1; m < 16; m <<= 1) zmax = fmaxf(zmax, __shfl_xor(zmax, m, 32));
  float zs = expf(zv - zmax);
#pragma unroll
  for (int m = 1; m < 16; m <<= 1) zs += __shfl_xor(zs, m, 32);
  float Z = zmax + logf(zs);

  // gold score: strided over t, reduced across the wave
  float sacc = 0.0f, lacc = 0.0f;
  for (int t = l; t < TLEN; t += 32) {
    int yt = y[b * TLEN + t];
    int yp = (t == 0) ? START_TAG : y[b * TLEN + t - 1];
    float mt = mask[b * TLEN + t];
    sacc += (emis[((size_t)b * TLEN + t) * NTAGS + yt] + tr[yt * NTAGS + yp]) * mt;
    lacc += mt;
  }
#pragma unroll
  for (int m = 1; m < 32; m <<= 1) {
    sacc += __shfl_xor(sacc, m, 32);
    lacc += __shfl_xor(lacc, m, 32);
  }
  if (l == 0) {
    int len = (int)(lacc + 0.5f);
    int last = (len == 0) ? START_TAG : y[b * TLEN + len - 1];
    float score = sacc + tr[STOP_TAG * NTAGS + last];
    out[b] = Z - score;
  }
}

// ---------------- host launcher ----------------

extern "C" void kernel_launch(void* const* d_in, const int* in_sizes, int n_in,
                              void* d_out, int out_size, void* d_ws, size_t ws_size,
                              hipStream_t stream) {
  const int*   x      = (const int*)d_in[0];
  const int*   y      = (const int*)d_in[1];
  const float* embedw = (const float*)d_in[2];
  const float* Wih0   = (const float*)d_in[3];
  const float* Whh0   = (const float*)d_in[4];
  const float* bih0   = (const float*)d_in[5];
  const float* bhh0   = (const float*)d_in[6];
  const float* Wih1   = (const float*)d_in[7];
  const float* Whh1   = (const float*)d_in[8];
  const float* bih1   = (const float*)d_in[9];
  const float* bhh1   = (const float*)d_in[10];
  const float* Wout   = (const float*)d_in[11];
  const float* bout   = (const float*)d_in[12];
  const float* trans  = (const float*)d_in[13];
  float* out = (float*)d_out;

  // workspace carve-up
  char* p = (char*)d_ws;
  const size_t XP_B   = (size_t)2 * TLEN * BATCH * NP * sizeof(float);      // 167.8 MB
  const size_t EH_B   = (size_t)TLEN * BATCH * KPX * sizeof(_Float16);      // 21.0 MB
  const size_t MASK_B = (size_t)BATCH * TLEN * sizeof(float);
  const size_t EMIS_B = (size_t)BATCH * TLEN * NTAGS * sizeof(float);
  const size_t WIH_B  = (size_t)2 * NP * KPX * sizeof(_Float16);
  const size_t WHH_B  = (size_t)2 * NP * KPH * sizeof(_Float16);
  const size_t WO_B   = (size_t)NTAGS * KPX * sizeof(_Float16);
  const size_t BP_B   = (size_t)2 * NP * sizeof(float);

  float*     XP    = (float*)p;            p += XP_B;
  _Float16*  e_h   = (_Float16*)p;         p += EH_B;
  _Float16*  h1    = (_Float16*)p;         p += EH_B;
  _Float16*  h2    = (_Float16*)p;         p += EH_B;
  float*     maskb = (float*)p;            p += MASK_B;
  float*     emis  = (float*)p;            p += EMIS_B;
  _Float16*  wih0p = (_Float16*)p;         p += WIH_B;
  _Float16*  whh0p = (_Float16*)p;         p += WHH_B;
  _Float16*  wih1p = (_Float16*)p;         p += WIH_B;
  _Float16*  whh1p = (_Float16*)p;         p += WHH_B;
  _Float16*  woutp = (_Float16*)p;         p += WO_B;
  float*     b0p   = (float*)p;            p += BP_B;
  float*     b1p   = (float*)p;            p += BP_B;

  const long EH_N = (long)TLEN * BATCH * KPX;      // 10,485,760
  const int ZB = (int)((EH_N + 255) / 256);
  const size_t WSH_DYN = (size_t)NP * KPH * sizeof(_Float16);  // 204800 B

  // prep
  zero_half_kernel<<<ZB, 256, 0, stream>>>(h1, EH_N);
  zero_half_kernel<<<ZB, 256, 0, stream>>>(h2, EH_N);
  pack_w_gates_kernel<<<(2 * NP * KPX + 255) / 256, 256, 0, stream>>>(Wih0, wih0p, EMB, KPX, 2 * NP * KPX);
  pack_w_gates_kernel<<<(2 * NP * KPH + 255) / 256, 256, 0, stream>>>(Whh0, whh0p, HALF, KPH, 2 * NP * KPH);
  pack_w_gates_kernel<<<(2 * NP * KPX + 255) / 256, 256, 0, stream>>>(Wih1, wih1p, EMB, KPX, 2 * NP * KPX);
  pack_w_gates_kernel<<<(2 * NP * KPH + 255) / 256, 256, 0, stream>>>(Whh1, whh1p, HALF, KPH, 2 * NP * KPH);
  pack_b_gates_kernel<<<(2 * NP + 255) / 256, 256, 0, stream>>>(bih0, bhh0, b0p);
  pack_b_gates_kernel<<<(2 * NP + 255) / 256, 256, 0, stream>>>(bih1, bhh1, b1p);
  pack_wout_kernel<<<(NTAGS * KPX + 255) / 256, 256, 0, stream>>>(Wout, woutp);
  embed_mask_kernel<<<ZB, 256, 0, stream>>>(x, embedw, e_h, maskb);

  // layer 0
  xproj_kernel<<<dim3(TLEN, 2), 256, 0, stream>>>(e_h, wih0p, b0p, XP);
  lstm_kernel<<<2, 256, WSH_DYN, stream>>>(XP, whh0p, maskb, h1);
  // layer 1 (XP reused)
  xproj_kernel<<<dim3(TLEN, 2), 256, 0, stream>>>(h1, wih1p, b1p, XP);
  lstm_kernel<<<2, 256, WSH_DYN, stream>>>(XP, whh1p, maskb, h2);

  // emissions + CRF
  emis_kernel<<<256, 256, 0, stream>>>(h2, woutp, bout, maskb, emis);
  crf_kernel<<<BATCH, 32, 0, stream>>>(emis, maskb, trans, y, out);

  (void)in_sizes; (void)n_in; (void)out_size; (void)ws_size;
}
